// CrossAttention_73461120630938
// MI455X (gfx1250) — matrix operand order repaired and checked by reading, not compiled, not run on hardware
//
#include <hip/hip_runtime.h>

// ---------------------------------------------------------------------------
// CDNA5 (gfx1250) wave32 WMMA bf16 implementation of windowed global attention
// with decomposed relative position bias (ViTDet/SAM style).
// B=8, H=W=32, L=1024, C=768, nh=12, hd=64.
//
// V stored d-major so all attention tiles are contiguous bf16 global->LDS
// copies, routed through GLOBAL_LOAD_ASYNC_TO_LDS_B32 (ASYNCcnt) when the
// toolchain exposes the builtin (probe: it does; param type is typed int*).
// ---------------------------------------------------------------------------

typedef __attribute__((ext_vector_type(16))) __bf16 v16bf;
typedef __attribute__((ext_vector_type(8)))  float  v8f;

union Frag16 { v16bf v; unsigned u[8]; };

#if defined(__has_builtin)
#if __has_builtin(__builtin_amdgcn_global_load_async_to_lds_b32)
#define HAVE_ASYNC_LDS 1
#endif
#endif
#ifndef HAVE_ASYNC_LDS
#define HAVE_ASYNC_LDS 0
#endif

#if HAVE_ASYNC_LDS
typedef __attribute__((address_space(1))) int gas_int;  // global
typedef __attribute__((address_space(3))) int las_int;  // LDS
#endif

// Copy one DWORD global -> LDS. Async path uses per-lane VGPR addresses on
// both sides and is tracked with ASYNCcnt (no VGPR data round-trip).
__device__ __forceinline__ void async_copy_u32(const unsigned* g, unsigned* l) {
#if HAVE_ASYNC_LDS
  __builtin_amdgcn_global_load_async_to_lds_b32(
      (gas_int*)(unsigned long long)(const void*)g,
      (las_int*)l,
      /*offset=*/0, /*cpol=*/0);
#else
  *l = *g;
#endif
}
__device__ __forceinline__ void async_wait_all() {
#if HAVE_ASYNC_LDS
  asm volatile("s_wait_asynccnt 0" ::: "memory");
#endif
}

__device__ __forceinline__ unsigned short f2bf(float x) {
  unsigned b = __float_as_uint(x);
  b += 0x7FFFu + ((b >> 16) & 1u);          // round-to-nearest-even
  return (unsigned short)(b >> 16);
}
__device__ __forceinline__ unsigned pack2(float a, float b) {
  return (unsigned)f2bf(a) | ((unsigned)f2bf(b) << 16);
}
__device__ __forceinline__ float bf2f(unsigned short h) {
  return __uint_as_float(((unsigned)h) << 16);
}

__device__ __forceinline__ v8f wmma_bf16(const Frag16& a, const Frag16& b, v8f c) {
  return __builtin_amdgcn_wmma_f32_16x16x32_bf16(
      /*neg_a=*/false, a.v, /*neg_b=*/false, b.v,
      /*c_mod=*/(short)0, c, /*reuse_a=*/false, /*reuse_b=*/false);
}

// A fragment: 16x32 bf16 (MxK). LDS layout: row-major [m][k] bf16 pairs,
// `base` points at (row 0, k=0) with uint granularity; `stride` in uints.
// Lanes 0-15 row=lane hold K 0..15; lanes 16-31 hold K 16..31 (matches B frag).
__device__ __forceinline__ Frag16 load_a_frag(const unsigned* base, int stride) {
  Frag16 f;
  const int lane = threadIdx.x & 31;
  const unsigned* r = base + (lane & 15) * stride;
  const int kb2 = (lane < 16) ? 0 : 8;      // (K base)/2
#pragma unroll
  for (int j = 0; j < 4; ++j) f.u[j] = r[kb2 + j];
#pragma unroll
  for (int j = 0; j < 4; ++j) f.u[4 + j] = r[kb2 + 4 + j];
  return f;
}

// B fragment: 32x16 bf16 (KxN). LDS layout: [n][k] bf16 pairs (k contiguous).
// Contiguous k order (same as A frag): lane=n (mod 16); lanes 0-15 hold K 0..15, lanes 16-31 hold K 16..31.
__device__ __forceinline__ Frag16 load_b_frag(const unsigned* base, int stride) {
  Frag16 f;
  const int lane = threadIdx.x & 31;
  const unsigned* r = base + (lane & 15) * stride + ((lane < 16) ? 0 : 8);
#pragma unroll
  for (int j = 0; j < 8; ++j) f.u[j] = r[j];
  return f;
}

// ---------------------------------------------------------------------------
// Tiled GEMM, 128x64 tile per 256-thread block (8 waves, 32x32 each).
// MODE 0: A = f32 x, out -> bf16 q/k (token-major) and v (d-major) ws (+bias).
// MODE 1: A = bf16 workspace activations (async-copied), out -> f32 (+bias).
// ---------------------------------------------------------------------------
template <int MODE>
__global__ __launch_bounds__(256) void gemm_bf16_kernel(
    const float* __restrict__ Af, const unsigned short* __restrict__ Abf,
    const float* __restrict__ W, const float* __restrict__ bias,
    unsigned short* __restrict__ qo, unsigned short* __restrict__ ko,
    unsigned short* __restrict__ vo, float* __restrict__ fo,
    int N, int K) {
  __shared__ unsigned lA[128 * 17];  // [m][kpair], padded stride: no bank conflicts
  __shared__ unsigned lB[64 * 17];   // [n][kpair]

  const int tid = threadIdx.x;
  const int lane = tid & 31;
  const int wave = tid >> 5;
  const int wm = wave >> 1;          // 0..3
  const int wn = wave & 1;           // 0..1
  const int rowBase = blockIdx.y * 128;
  const int colBase = blockIdx.x * 64;

  v8f acc[2][2];
  v8f vzero = {};
#pragma unroll
  for (int i = 0; i < 2; ++i)
#pragma unroll
    for (int j = 0; j < 2; ++j) acc[i][j] = vzero;

  const unsigned* Au = (const unsigned*)Abf;
  const int Kw = K >> 1;

  for (int k0 = 0; k0 < K; k0 += 32) {
    __syncthreads();
    // stage A tile 128x32 bf16 (async when A is already bf16)
    for (int idx = tid; idx < 128 * 16; idx += 256) {
      int r = idx >> 4, kp = idx & 15;
      if (MODE == 0) {
        const float* ap = Af + (size_t)(rowBase + r) * K + (k0 + kp * 2);
        lA[r * 17 + kp] = pack2(ap[0], ap[1]);
      } else {
        async_copy_u32(Au + (size_t)(rowBase + r) * Kw + (k0 >> 1) + kp,
                       &lA[r * 17 + kp]);
      }
    }
    // stage B tile 32x64 (transposed to [n][k]) f32 -> bf16
    for (int idx = tid; idx < 64 * 16; idx += 256) {
      int n = idx >> 4, kp = idx & 15;
      const float* wp = W + (size_t)(k0 + kp * 2) * N + (colBase + n);
      lB[n * 17 + kp] = pack2(wp[0], wp[N]);
    }
    if (MODE != 0) async_wait_all();
    __syncthreads();

    Frag16 a0 = load_a_frag(lA + (wm * 32) * 17, 17);
    Frag16 a1 = load_a_frag(lA + (wm * 32 + 16) * 17, 17);
    Frag16 b0 = load_b_frag(lB + (wn * 32) * 17, 17);
    Frag16 b1 = load_b_frag(lB + (wn * 32 + 16) * 17, 17);
    acc[0][0] = wmma_bf16(a0, b0, acc[0][0]);
    acc[0][1] = wmma_bf16(a0, b1, acc[0][1]);
    acc[1][0] = wmma_bf16(a1, b0, acc[1][0]);
    acc[1][1] = wmma_bf16(a1, b1, acc[1][1]);
  }

  const int half8 = (lane < 16) ? 0 : 8;
  const int cLane = lane & 15;
#pragma unroll
  for (int mt = 0; mt < 2; ++mt) {
#pragma unroll
    for (int nt = 0; nt < 2; ++nt) {
#pragma unroll
      for (int i = 0; i < 8; ++i) {
        int row = rowBase + wm * 32 + mt * 16 + i + half8;
        int col = colBase + wn * 32 + nt * 16 + cLane;
        float val = acc[mt][nt][i] + bias[col];
        if (MODE == 0) {
          int which = col / 768;
          int cc = col - which * 768;
          int head = cc >> 6, d = cc & 63;
          int bb = row >> 10, l = row & 1023;
          int bhh = bb * 12 + head;
          unsigned short hv = f2bf(val);
          if (which == 0)      qo[((size_t)bhh * 1024 + l) * 64 + d] = hv;
          else if (which == 1) ko[((size_t)bhh * 1024 + l) * 64 + d] = hv;
          else                 vo[((size_t)bhh * 64 + d) * 1024 + l] = hv;  // d-major
        } else {
          fo[(size_t)row * N + col] = val;
        }
      }
    }
  }
}

// ---------------------------------------------------------------------------
// Fused flash attention with decomposed rel-pos bias.
// grid = (L/64 q-tiles, B*nh). 128 threads = 4 waves; wave w owns q-rows
// [16w, 16w+16) of the tile -> online-softmax stats are wave-local.
// Q/K/V tiles stream global->LDS through the async path (ASYNCcnt).
// ---------------------------------------------------------------------------
__global__ __launch_bounds__(128) void attn_kernel(
    const unsigned short* __restrict__ qb, const unsigned short* __restrict__ kb,
    const unsigned short* __restrict__ vb,
    const float* __restrict__ rph, const float* __restrict__ rpw,
    unsigned short* __restrict__ out) {
  __shared__ unsigned sQ[64 * 33];      // Q tile [64 q-rows][64 d] bf16
  __shared__ unsigned sK[64 * 33];      // K tile [64 k-rows][64 d] bf16
  __shared__ unsigned sVt[64 * 33];     // V tile [64 d][64 k-rows] bf16 (d-major ws)
  __shared__ float    sRelH[64 * 32];   // [q row in tile][kh]
  __shared__ float    sRelW[64 * 32];   // [q row in tile][kw]
  __shared__ unsigned short sP[4 * 16 * 66];  // per-wave P strip [16][64] bf16

  const int tid = threadIdx.x;
  const int lane = tid & 31;
  const int wave = tid >> 5;
  const int bh = blockIdx.y;                  // b*12 + head
  const int b = bh / 12, head = bh % 12;
  const int qt = blockIdx.x;
  const float scale = 0.125f;                 // 64^-0.5

  const unsigned* qg = (const unsigned*)qb;
  const unsigned* kg = (const unsigned*)kb;
  const unsigned* vg = (const unsigned*)vb;
  const size_t base_u = (size_t)bh * 1024 * 32;  // q/k: uint rows of 32 per (b,head)
  const size_t vbase  = (size_t)bh * 64 * 512;   // v:   uint rows of 512 (d-major)

  // --- load Q tile (async global->LDS) ---
  for (int idx = tid; idx < 64 * 32; idx += 128) {
    int r = idx >> 5, dp = idx & 31;
    async_copy_u32(qg + base_u + (size_t)(qt * 64 + r) * 32 + dp, &sQ[r * 33 + dp]);
  }
  async_wait_all();
  __syncthreads();

  // --- rel-pos bias tables: relH[m][kh] = q[m,:]·rel_pos_h[hq-kh+31,:] ---
  for (int idx = tid; idx < 64 * 32; idx += 128) {
    int m = idx >> 5, kk = idx & 31;
    int mg = qt * 64 + m;
    int hq = mg >> 5, wq = mg & 31;
    const float* rh = rph + (size_t)(hq - kk + 31) * 64;
    const float* rw = rpw + (size_t)(wq - kk + 31) * 64;
    const unsigned* qr = sQ + m * 33;
    float aH = 0.f, aW = 0.f;
    for (int dp = 0; dp < 32; ++dp) {
      unsigned u = qr[dp];
      float q0 = bf2f((unsigned short)(u & 0xFFFFu));
      float q1 = bf2f((unsigned short)(u >> 16));
      aH += q0 * rh[2 * dp] + q1 * rh[2 * dp + 1];
      aW += q0 * rw[2 * dp] + q1 * rw[2 * dp + 1];
    }
    sRelH[idx] = aH;
    sRelW[idx] = aW;
  }
  __syncthreads();

  Frag16 aq0 = load_a_frag(sQ + wave * 16 * 33, 33);       // d = 0..31
  Frag16 aq1 = load_a_frag(sQ + wave * 16 * 33 + 16, 33);  // d = 32..63

  v8f oacc[4];
  v8f vzero = {};
#pragma unroll
  for (int dt = 0; dt < 4; ++dt) oacc[dt] = vzero;
  float mrun[8], lrun[8];
#pragma unroll
  for (int i = 0; i < 8; ++i) { mrun[i] = -3.0e38f; lrun[i] = 0.f; }

  const unsigned* sPu = (const unsigned*)sP;
  const int half8 = (lane < 16) ? 0 : 8;
  const int cLane = lane & 15;

  for (int kt = 0; kt < 16; ++kt) {
    __syncthreads();  // previous iteration done with sK/sVt
    // async stream K tile (token-major) and V tile (d-major) into LDS
    for (int idx = tid; idx < 64 * 32; idx += 128) {
      int r = idx >> 5, dp = idx & 31;
      async_copy_u32(kg + base_u + (size_t)(kt * 64 + r) * 32 + dp, &sK[r * 33 + dp]);
      async_copy_u32(vg + vbase + (size_t)r * 512 + kt * 32 + dp, &sVt[r * 33 + dp]);
    }
    if (kt + 1 < 16)  // warm GL2 for next K tile (global_prefetch_b8)
      __builtin_prefetch(kg + base_u + (size_t)((kt + 1) * 64) * 32, 0, 1);
    async_wait_all();
    __syncthreads();

    // S = scale * Q K^T + relH + relW
    float pv[4][8];
#pragma unroll
    for (int nt = 0; nt < 4; ++nt) {
      Frag16 bk0 = load_b_frag(sK + nt * 16 * 33, 33);
      Frag16 bk1 = load_b_frag(sK + nt * 16 * 33 + 16, 33);
      v8f s = vzero;
      s = wmma_bf16(aq0, bk0, s);
      s = wmma_bf16(aq1, bk1, s);
      int kkcol = kt * 64 + nt * 16 + cLane;
      int kh = kkcol >> 5, kw = kkcol & 31;
#pragma unroll
      for (int i = 0; i < 8; ++i) {
        int mloc = wave * 16 + i + half8;
        pv[nt][i] = s[i] * scale + sRelH[mloc * 32 + kh] + sRelW[mloc * 32 + kw];
      }
    }

    // online softmax: each C-matrix row lives in one 16-lane half
#pragma unroll
    for (int i = 0; i < 8; ++i) {
      float rmax = fmaxf(fmaxf(pv[0][i], pv[1][i]), fmaxf(pv[2][i], pv[3][i]));
#pragma unroll
      for (int off = 8; off >= 1; off >>= 1)
        rmax = fmaxf(rmax, __shfl_xor(rmax, off, 16));
      float mnew = fmaxf(mrun[i], rmax);
      float corr = __expf(mrun[i] - mnew);
      float rsum = 0.f;
#pragma unroll
      for (int nt = 0; nt < 4; ++nt) {
        float p = __expf(pv[nt][i] - mnew);
        pv[nt][i] = p;
        rsum += p;
      }
#pragma unroll
      for (int off = 8; off >= 1; off >>= 1)
        rsum += __shfl_xor(rsum, off, 16);
      lrun[i] = lrun[i] * corr + rsum;
      mrun[i] = mnew;
#pragma unroll
      for (int dt = 0; dt < 4; ++dt) oacc[dt][i] *= corr;
    }

    // C-frag -> A-frag relayout through this wave's private LDS strip
#pragma unroll
    for (int nt = 0; nt < 4; ++nt)
#pragma unroll
      for (int i = 0; i < 8; ++i)
        sP[(wave * 16 + i + half8) * 66 + nt * 16 + cLane] = f2bf(pv[nt][i]);
    asm volatile("s_wait_dscnt 0" ::: "memory");  // wave-local LDS RAW fence

    Frag16 ap0 = load_a_frag(sPu + wave * 16 * 33, 33);       // k = 0..31
    Frag16 ap1 = load_a_frag(sPu + wave * 16 * 33 + 16, 33);  // k = 32..63
#pragma unroll
    for (int dt = 0; dt < 4; ++dt) {
      Frag16 bv0 = load_b_frag(sVt + dt * 16 * 33, 33);
      Frag16 bv1 = load_b_frag(sVt + dt * 16 * 33 + 16, 33);
      oacc[dt] = wmma_bf16(ap0, bv0, oacc[dt]);
      oacc[dt] = wmma_bf16(ap1, bv1, oacc[dt]);
    }
  }

  // finalize: O /= l ; scatter to [B, L, C] bf16 (head-interleaved)
#pragma unroll
  for (int dt = 0; dt < 4; ++dt) {
#pragma unroll
    for (int i = 0; i < 8; ++i) {
      int l = qt * 64 + wave * 16 + i + half8;
      int d = dt * 16 + cLane;
      float val = oacc[dt][i] / lrun[i];
      out[((size_t)b * 1024 + l) * 768 + head * 64 + d] = f2bf(val);
    }
  }
}

// ---------------------------------------------------------------------------
extern "C" void kernel_launch(void* const* d_in, const int* in_sizes, int n_in,
                              void* d_out, int out_size, void* d_ws, size_t ws_size,
                              hipStream_t stream) {
  (void)in_sizes; (void)n_in; (void)out_size; (void)ws_size;
  const float* x      = (const float*)d_in[0];  // [8,32,32,768]
  const float* qkv_w  = (const float*)d_in[1];  // [768,2304]
  const float* qkv_b  = (const float*)d_in[2];  // [2304]
  const float* proj_w = (const float*)d_in[3];  // [768,768]
  const float* proj_b = (const float*)d_in[4];  // [768]
  const float* rph    = (const float*)d_in[5];  // [63,64]
  const float* rpw    = (const float*)d_in[6];  // [63,64]
  float* out = (float*)d_out;                   // [8,32,32,768]

  // bf16 workspace: q,k [B,nh,L,64]; v [B,nh,64,L]; attn output [B,L,C]
  const size_t HEADS_BYTES = (size_t)8 * 12 * 1024 * 64 * 2;  // 12.6 MB each
  char* ws = (char*)d_ws;
  unsigned short* qws = (unsigned short*)(ws);
  unsigned short* kws = (unsigned short*)(ws + HEADS_BYTES);
  unsigned short* vws = (unsigned short*)(ws + 2 * HEADS_BYTES);
  unsigned short* aws = (unsigned short*)(ws + 3 * HEADS_BYTES);

  // 1) QKV projection: [8192,768] x [768,2304]
  dim3 g1(2304 / 64, 8192 / 128);
  gemm_bf16_kernel<0><<<g1, 256, 0, stream>>>(x, nullptr, qkv_w, qkv_b,
                                              qws, kws, vws, nullptr,
                                              2304, 768);
  // 2) Fused attention + rel-pos bias + softmax + P·V
  dim3 g2(1024 / 64, 8 * 12);
  attn_kernel<<<g2, 128, 0, stream>>>(qws, kws, vws, rph, rpw, aws);

  // 3) Output projection: [8192,768] x [768,768] -> f32 out
  dim3 g3(768 / 64, 8192 / 128);
  gemm_bf16_kernel<1><<<g3, 256, 0, stream>>>(nullptr, aws, proj_w, proj_b,
                                              nullptr, nullptr, nullptr, out,
                                              768, 768);
}
